// WaveNetModel_49443663511773
// MI455X (gfx1250) — compile-verified, matrix operand order checked by reading, not run on hardware
//
// CDNA5 / gfx1250 WaveNet: 40 dilated layers as bf16 WMMA GEMMs, f32 accum.
// Windows shrink per layer (only the receptive field of the last 512 outputs
// is computed). Weights pre-packed into WMMA A-fragment lane order (bf16).
#include <hip/hip_runtime.h>
#include <stdint.h>

typedef __attribute__((ext_vector_type(16))) __bf16 v16bf;
typedef __attribute__((ext_vector_type(8)))  float  v8f;
typedef __attribute__((ext_vector_type(4)))  unsigned int u32x4;

union Frag { v16bf v; u32x4 q[2]; };

__device__ __forceinline__ unsigned short f2bf(float f) {
    union { float f; unsigned int u; } a; a.f = f;
    unsigned int r = a.u + 0x7FFFu + ((a.u >> 16) & 1u);   // round-nearest-even
    return (unsigned short)(r >> 16);
}
__device__ __forceinline__ float bf2f(unsigned short s) {
    union { unsigned int u; float f; } a; a.u = ((unsigned int)s) << 16;
    return a.f;
}
__device__ __forceinline__ v8f wmma_bf16(const Frag& a, const Frag& b, v8f c) {
    return __builtin_amdgcn_wmma_f32_16x16x32_bf16(false, a.v, false, b.v,
                                                   (short)0, c, false, false);
}

// K position inside a 32-wide k-step for fragment element e on a given lane,
// per ISA 7.12.2 (16-bit matrices, wave32).
__device__ __forceinline__ int koff_of(int lane, int e) {
    return ((lane < 16) ? 0 : 8) + ((e < 8) ? e : e + 8);
}

// ---------------------------------------------------------------- weight pack
__global__ void __launch_bounds__(256)
wavenet_pack(const float* __restrict__ dil,  // (40,256,128,2)
             const float* __restrict__ rsw,  // (40,128,128,1)
             const float* __restrict__ skw,  // (40,512,128,1)
             const float* __restrict__ enw,  // (256,512,1)
             unsigned short* __restrict__ wd,
             unsigned short* __restrict__ wr,
             unsigned short* __restrict__ wsk,
             unsigned short* __restrict__ we)
{
    const long long stride = (long long)gridDim.x * blockDim.x;
    const long long base   = (long long)blockIdx.x * blockDim.x + threadIdx.x;

    // dilated: 40 layers x 16 Mtiles x 8 ksteps x 32 lanes x 16 halves
    for (long long i = base; i < 40LL * 16 * 8 * 512; i += stride) {
        const int e = (int)(i & 15), lane = (int)((i >> 4) & 31);
        const long long rest = i >> 9;
        const int kk = (int)(rest & 7), mt = (int)((rest >> 3) & 15);
        const int layer = (int)(rest >> 7);
        const int M = mt * 16 + (lane & 15);
        const int K = kk * 32 + koff_of(lane, e);
        const int tap = K >> 7, ch = K & 127;   // K = [tap0 ch0..127 | tap1 ch0..127]
        wd[i] = f2bf(dil[(((size_t)layer * 256 + M) * 128 + ch) * 2 + tap]);
    }
    // residual: 40 x 8 x 4 x 512
    for (long long i = base; i < 40LL * 8 * 4 * 512; i += stride) {
        const int e = (int)(i & 15), lane = (int)((i >> 4) & 31);
        const long long rest = i >> 9;
        const int kk = (int)(rest & 3), mt = (int)((rest >> 2) & 7);
        const int layer = (int)(rest >> 5);
        const int M = mt * 16 + (lane & 15);
        const int K = kk * 32 + koff_of(lane, e);
        wr[i] = f2bf(rsw[((size_t)layer * 128 + M) * 128 + K]);
    }
    // skip: 40 x 32 x 4 x 512
    for (long long i = base; i < 40LL * 32 * 4 * 512; i += stride) {
        const int e = (int)(i & 15), lane = (int)((i >> 4) & 31);
        const long long rest = i >> 9;
        const int kk = (int)(rest & 3), mt = (int)((rest >> 2) & 31);
        const int layer = (int)(rest >> 7);
        const int M = mt * 16 + (lane & 15);
        const int K = kk * 32 + koff_of(lane, e);
        wsk[i] = f2bf(skw[((size_t)layer * 512 + M) * 128 + K]);
    }
    // end: 16 x 16 x 512
    for (long long i = base; i < 16LL * 16 * 512; i += stride) {
        const int e = (int)(i & 15), lane = (int)((i >> 4) & 31);
        const long long rest = i >> 9;
        const int kk = (int)(rest & 15), mt = (int)(rest >> 4);
        const int M = mt * 16 + (lane & 15);
        const int K = kk * 32 + koff_of(lane, e);
        we[i] = f2bf(enw[(size_t)M * 512 + K]);
    }
}

// ---------------------------------------------------------------- start conv
__global__ void __launch_bounds__(256)
wavenet_start(const float* __restrict__ in,   // (4,1,8192)
              const float* __restrict__ sw,   // (128,1,1)
              unsigned short* __restrict__ x0)
{
    const long long total  = 4LL * 128 * 8192;
    const long long stride = (long long)gridDim.x * blockDim.x;
    for (long long i = (long long)blockIdx.x * blockDim.x + threadIdx.x;
         i < total; i += stride) {
        const int t = (int)(i & 8191);
        const int c = (int)((i >> 13) & 127);
        const int n = (int)(i >> 20);
        x0[i] = f2bf(sw[c] * in[(size_t)n * 8192 + t]);
    }
}

// ---------------------------------------------------------------- layer
__global__ void __launch_bounds__(256)
wavenet_layer(const unsigned short* __restrict__ xin,
              unsigned short* __restrict__ xout,
              const unsigned short* __restrict__ wd,   // this layer's frags
              const unsigned short* __restrict__ wsk,
              const unsigned short* __restrict__ wr,
              float* __restrict__ skip_acc,            // (4,512,512)
              int d, int t_base, int first_layer)
{
    __shared__ __attribute__((aligned(16))) unsigned short actB[2][16][128];
    __shared__ __attribute__((aligned(16))) float          hbuf[256][16];
    __shared__ __attribute__((aligned(16))) unsigned short zbuf[16][128];

    const int tid  = threadIdx.x;
    const int lane = tid & 31;
    const int wave = tid >> 5;
    const int n    = blockIdx.y;
    const int t0   = t_base + (int)blockIdx.x * 16;
    const size_t xbase = (size_t)n * 128 * 8192;

    // ---- stage both input taps (bf16) into LDS as [tap][t][ch] ----
    {
        const int ch = tid >> 1;
        const int th = tid & 1;
        const unsigned short* s0 = xin + xbase + (size_t)ch * 8192 + (t0 - d + th * 8);
        #pragma unroll
        for (int j = 0; j < 8; ++j) actB[0][th * 8 + j][ch] = s0[j];
        const u32x4 vv = *(const u32x4*)(xin + xbase + (size_t)ch * 8192 + (t0 + th * 8));
        const unsigned short* s1 = (const unsigned short*)&vv;
        #pragma unroll
        for (int j = 0; j < 8; ++j) actB[1][th * 8 + j][ch] = s1[j];
    }
    __syncthreads();

    const int tcol = lane & 15;
    const int khi  = (lane < 16) ? 0 : 8;

    // ---- dilated conv: 256x16 out, K=256; wave covers 2 M-tiles ----
    v8f acc0 = {0,0,0,0,0,0,0,0};
    v8f acc1 = {0,0,0,0,0,0,0,0};
    const int mt0 = wave * 2, mt1 = mt0 + 1;
    #pragma unroll
    for (int kk = 0; kk < 8; ++kk) {
        const int tap = kk >> 2;
        const int kb  = (kk & 3) * 32 + khi;
        Frag b;
        b.q[0] = *(const u32x4*)&actB[tap][tcol][kb];
        b.q[1] = *(const u32x4*)&actB[tap][tcol][kb + 16];
        Frag a0, a1;
        const u32x4* A0 = (const u32x4*)(wd + ((size_t)(mt0 * 8 + kk) * 32 + lane) * 16);
        const u32x4* A1 = (const u32x4*)(wd + ((size_t)(mt1 * 8 + kk) * 32 + lane) * 16);
        a0.q[0] = A0[0]; a0.q[1] = A0[1];
        a1.q[0] = A1[0]; a1.q[1] = A1[1];
        acc0 = wmma_bf16(a0, b, acc0);
        acc1 = wmma_bf16(a1, b, acc1);
    }
    #pragma unroll
    for (int r = 0; r < 8; ++r) {
        hbuf[mt0 * 16 + r + khi][tcol] = acc0[r];
        hbuf[mt1 * 16 + r + khi][tcol] = acc1[r];
    }
    __syncthreads();

    // ---- gate: z = tanh(h0) * sigmoid(h1), bf16 into [t][ch] ----
    #pragma unroll
    for (int j = 0; j < 8; ++j) {
        const int idx = tid * 8 + j;
        const int c = idx >> 4, tt = idx & 15;
        const float f = hbuf[c][tt];
        const float g = hbuf[c + 128][tt];
        zbuf[tt][c] = f2bf(tanhf(f) * (1.0f / (1.0f + __expf(-g))));
    }
    __syncthreads();

    // ---- residual 1x1: 128x16, wave covers 1 M-tile ----
    v8f racc = {0,0,0,0,0,0,0,0};
    #pragma unroll
    for (int kk = 0; kk < 4; ++kk) {
        const int kb = kk * 32 + khi;
        Frag b;
        b.q[0] = *(const u32x4*)&zbuf[tcol][kb];
        b.q[1] = *(const u32x4*)&zbuf[tcol][kb + 16];
        Frag a;
        const u32x4* A = (const u32x4*)(wr + ((size_t)(wave * 4 + kk) * 32 + lane) * 16);
        a.q[0] = A[0]; a.q[1] = A[1];
        racc = wmma_bf16(a, b, racc);
    }
    {
        const int tcur = t0 + tcol;
        #pragma unroll
        for (int r = 0; r < 8; ++r) {
            const int m = wave * 16 + r + khi;
            const size_t o = xbase + (size_t)m * 8192 + tcur;
            xout[o] = f2bf(racc[r] + bf2f(xin[o]));
        }
    }

    // ---- skip 1x1 on the trailing 512 steps: 512x16, wave covers 4 M-tiles ----
    if (t0 >= 8192 - 512) {
        v8f sa[4];
        #pragma unroll
        for (int u = 0; u < 4; ++u) sa[u] = (v8f){0,0,0,0,0,0,0,0};
        #pragma unroll
        for (int kk = 0; kk < 4; ++kk) {
            const int kb = kk * 32 + khi;
            Frag b;
            b.q[0] = *(const u32x4*)&zbuf[tcol][kb];
            b.q[1] = *(const u32x4*)&zbuf[tcol][kb + 16];
            #pragma unroll
            for (int u = 0; u < 4; ++u) {
                Frag a;
                const u32x4* A = (const u32x4*)
                    (wsk + ((size_t)((wave * 4 + u) * 4 + kk) * 32 + lane) * 16);
                a.q[0] = A[0]; a.q[1] = A[1];
                sa[u] = wmma_bf16(a, b, sa[u]);
            }
        }
        const int ts = t0 - (8192 - 512) + tcol;
        #pragma unroll
        for (int u = 0; u < 4; ++u) {
            #pragma unroll
            for (int r = 0; r < 8; ++r) {
                const int m = (wave * 4 + u) * 16 + r + khi;
                const size_t o = (size_t)n * 512 * 512 + (size_t)m * 512 + ts;
                skip_acc[o] = first_layer ? sa[u][r] : (skip_acc[o] + sa[u][r]);
            }
        }
    }
}

// ---------------------------------------------------------------- end conv
__global__ void __launch_bounds__(256)
wavenet_end(const float* __restrict__ skip_acc,        // (4,512,512)
            const unsigned short* __restrict__ we,     // packed (256x512)
            const float* __restrict__ end_b,           // (256)
            float* __restrict__ out)                   // (4*512, 256)
{
    __shared__ __attribute__((aligned(16))) unsigned short sb[16][512];
    const int tid = threadIdx.x, lane = tid & 31, wave = tid >> 5;
    const int n = blockIdx.y;
    const int tloc = (int)blockIdx.x * 16;

    #pragma unroll
    for (int rep = 0; rep < 2; ++rep) {
        const int ch = tid + rep * 256;
        const float* p = skip_acc + ((size_t)n * 512 + ch) * 512 + tloc;
        #pragma unroll
        for (int j = 0; j < 16; ++j) {
            const float v = p[j];
            sb[j][ch] = f2bf(v > 0.0f ? v : 0.0f);
        }
    }
    __syncthreads();

    const int tcol = lane & 15;
    const int khi  = (lane < 16) ? 0 : 8;
    const int mt0 = wave * 2, mt1 = mt0 + 1;
    v8f c0 = {0,0,0,0,0,0,0,0};
    v8f c1 = {0,0,0,0,0,0,0,0};
    #pragma unroll
    for (int kk = 0; kk < 16; ++kk) {
        const int kb = kk * 32 + khi;
        Frag b;
        b.q[0] = *(const u32x4*)&sb[tcol][kb];
        b.q[1] = *(const u32x4*)&sb[tcol][kb + 16];
        Frag a0, a1;
        const u32x4* A0 = (const u32x4*)(we + ((size_t)(mt0 * 16 + kk) * 32 + lane) * 16);
        const u32x4* A1 = (const u32x4*)(we + ((size_t)(mt1 * 16 + kk) * 32 + lane) * 16);
        a0.q[0] = A0[0]; a0.q[1] = A0[1];
        a1.q[0] = A1[0]; a1.q[1] = A1[1];
        c0 = wmma_bf16(a0, b, c0);
        c1 = wmma_bf16(a1, b, c1);
    }
    const int row = n * 512 + tloc + tcol;
    #pragma unroll
    for (int r = 0; r < 8; ++r) {
        const int cc0 = mt0 * 16 + r + khi;
        const int cc1 = mt1 * 16 + r + khi;
        out[(size_t)row * 256 + cc0] = c0[r] + end_b[cc0];
        out[(size_t)row * 256 + cc1] = c1[r] + end_b[cc1];
    }
}

// ---------------------------------------------------------------- host
extern "C" void kernel_launch(void* const* d_in, const int* in_sizes, int n_in,
                              void* d_out, int out_size, void* d_ws, size_t ws_size,
                              hipStream_t stream)
{
    const float* input      = (const float*)d_in[0];
    const float* start_w    = (const float*)d_in[1];
    const float* dilated_w  = (const float*)d_in[2];
    const float* residual_w = (const float*)d_in[3];
    const float* skip_w     = (const float*)d_in[4];
    const float* end_w      = (const float*)d_in[5];
    const float* end_b      = (const float*)d_in[6];
    float* out = (float*)d_out;

    char* ws = (char*)d_ws;
    size_t off = 0;
    auto carve = [&](size_t bytes) -> void* {
        off = (off + 255) & ~(size_t)255;
        void* p = ws + off;
        off += bytes;
        return p;
    };
    unsigned short* x0  = (unsigned short*)carve(4ull * 128 * 8192 * 2);
    unsigned short* x1  = (unsigned short*)carve(4ull * 128 * 8192 * 2);
    float*        skipA = (float*)        carve(4ull * 512 * 512 * 4);
    unsigned short* wd  = (unsigned short*)carve(40ull * 16 * 8 * 512 * 2);
    unsigned short* wr  = (unsigned short*)carve(40ull * 8 * 4 * 512 * 2);
    unsigned short* wsk = (unsigned short*)carve(40ull * 32 * 4 * 512 * 2);
    unsigned short* we  = (unsigned short*)carve(16ull * 16 * 512 * 2);

    wavenet_pack <<<2048, 256, 0, stream>>>(dilated_w, residual_w, skip_w, end_w,
                                            wd, wr, wsk, we);
    wavenet_start<<<4096, 256, 0, stream>>>(input, start_w, x0);

    int dil[40];
    { int li = 0; for (int b = 0; b < 4; ++b) { int d = 1;
        for (int l = 0; l < 10; ++l) { dil[li++] = d; d *= 2; } } }
    int W[40];
    { int acc = 512; for (int i = 39; i >= 0; --i) { W[i] = acc; acc += dil[i]; } }

    unsigned short* xp[2] = { x0, x1 };
    for (int i = 0; i < 40; ++i) {
        const int tiles  = (W[i] + 15) / 16;
        const int t_base = 8192 - tiles * 16;
        dim3 grid(tiles, 4);
        wavenet_layer<<<grid, 256, 0, stream>>>(
            xp[i & 1], xp[(i + 1) & 1],
            wd  + (size_t)i * 16 * 8 * 512,
            wsk + (size_t)i * 32 * 4 * 512,
            wr  + (size_t)i * 8 * 4 * 512,
            skipA, dil[i], t_base, (i == 0) ? 1 : 0);
    }
    wavenet_end<<<dim3(32, 4), 256, 0, stream>>>(skipA, we, end_b, out);

    (void)in_sizes; (void)n_in; (void)out_size; (void)ws_size;
}
